// DGNN_61924838474297
// MI455X (gfx1250) — matrix operand order repaired and checked by reading, not compiled, not run on hardware
//
#include <hip/hip_runtime.h>
#include <hip/hip_bf16.h>
#include <math.h>

// ---------------------------------------------------------------------------
// Problem constants (match reference)
// ---------------------------------------------------------------------------
#define NN      20000     // nodes
#define NE      320000    // edges
#define ESL     (NE + NN) // edges incl. self loops
#define NG      256       // graphs
#define DMAX    512
#define NY      10

typedef __attribute__((ext_vector_type(16))) __bf16 v16bf;
typedef __attribute__((ext_vector_type(8)))  float  v8f;

// ---------------------------------------------------------------------------
// Weight pre-pack: W[K,N] fp32 -> bf16 fragment-ordered tiles.
// Tile (kt,nt) covers K rows [kt*32, kt*32+32) x N cols [nt*16, nt*16+16).
// Within a tile, lane L (half=L>>4, mn=L&15) stores its 16 bf16 B-fragment
// elements contiguously at tile_base + L*16 elements (32 bytes), i.e. exactly
// the v16bf the WMMA consumes: b[j] = W[kt*32 + 8*half + (j<8 ? j : j+8)][nt*16 + mn].
// ---------------------------------------------------------------------------
__global__ void pack_w_bf16(const float* __restrict__ W, __bf16* __restrict__ PW,
                            int K, int N)
{
    int i = blockIdx.x * blockDim.x + threadIdx.x;
    int total = (K >> 5) * (N >> 4) * 32;
    if (i >= total) return;
    int lane = i & 31, tile = i >> 5;
    int ntn = N >> 4;
    int kt = tile / ntn, nt = tile % ntn;
    int half = lane >> 4, mn = lane & 15;
    const float* wp = W + (size_t)(kt * 32 + 8 * half) * N + nt * 16 + mn;
    v16bf b;
    #pragma unroll
    for (int j = 0; j < 16; ++j)
        b[j] = (__bf16)wp[(size_t)(j < 8 ? j : j + 8) * N];
    *(v16bf*)(PW + (size_t)tile * 512 + lane * 16) = b;
}

// ---------------------------------------------------------------------------
// Fragment loaders for the pipelined GEMM
// ---------------------------------------------------------------------------
__device__ __forceinline__ v16bf load_a_frag(const float* __restrict__ ap)
{
    float4 a0 = *(const float4*)(ap);
    float4 a1 = *(const float4*)(ap + 4);
    float4 a2 = *(const float4*)(ap + 16);
    float4 a3 = *(const float4*)(ap + 20);
    v16bf a;
    a[0]  = (__bf16)a0.x; a[1]  = (__bf16)a0.y; a[2]  = (__bf16)a0.z; a[3]  = (__bf16)a0.w;
    a[4]  = (__bf16)a1.x; a[5]  = (__bf16)a1.y; a[6]  = (__bf16)a1.z; a[7]  = (__bf16)a1.w;
    a[8]  = (__bf16)a2.x; a[9]  = (__bf16)a2.y; a[10] = (__bf16)a2.z; a[11] = (__bf16)a2.w;
    a[12] = (__bf16)a3.x; a[13] = (__bf16)a3.y; a[14] = (__bf16)a3.z; a[15] = (__bf16)a3.w;
    return a;
}

__device__ __forceinline__ void load_b_frags(const __bf16* __restrict__ pb, v16bf* b)
{
    b[0] = *(const v16bf*)(pb);
    b[1] = *(const v16bf*)(pb + 512);
    b[2] = *(const v16bf*)(pb + 1024);
    b[3] = *(const v16bf*)(pb + 1536);
}

// ---------------------------------------------------------------------------
// WMMA bf16 GEMM:  C[M,N] = A[M,K] @ W[K,N]  (+bias) (+relu)
// One wave per 16x64 output strip (4 accumulators sharing one A fragment),
// 4 waves / block, software-pipelined K loop with two fragment register sets
// (K/32 is even for all layers, so the double-step loop is exact).
// Requires M%16==0, K%64==0, N%64==0, (M/16)*(N/64)%4==0 (all hold here).
// ---------------------------------------------------------------------------
__global__ __launch_bounds__(128)
void wmma_gemm_bf16(const float* __restrict__ A, const __bf16* __restrict__ PW,
                    const float* __restrict__ bias, float* __restrict__ C,
                    int M, int K, int N, int relu)
{
    const int lane = threadIdx.x & 31;
    const int wave = threadIdx.x >> 5;
    const int half = lane >> 4;     // 0: lanes 0-15, 1: lanes 16-31
    const int mn   = lane & 15;

    const int nchunks = N >> 6;                  // 64-col strips per row-tile
    const int tid  = blockIdx.x * 4 + wave;      // wave-strip id
    const int tm   = tid / nchunks;              // 16-row tile index
    const int tn0  = (tid % nchunks) * 64;       // first output column
    const int ntn  = N >> 4;                     // 16-col tiles per K-tile row

    v8f acc[4] = {{}, {}, {}, {}};
    const float*  arow = A + (size_t)(tm * 16 + mn) * K + 8 * half;
    const __bf16* pb   = PW + (size_t)(tn0 >> 4) * 512 + lane * 16;
    const size_t  pbst = (size_t)ntn * 512;      // packed stride per K-tile

    if (bias) __builtin_prefetch(bias + tn0, 0, 3);   // gfx1250 global_prefetch

    // ---- software pipeline: two fragment sets, no loop-carried copies ----
    v16bf a0f, a1f, b0f[4], b1f[4];
    a0f = load_a_frag(arow);
    load_b_frags(pb, b0f);

    int k0 = 0;
    for (; k0 < K - 64; k0 += 64) {
        // stage 0: issue loads for k0+32, compute on set0 (k0)
        a1f = load_a_frag(arow + k0 + 32);
        load_b_frags(pb + ((size_t)((k0 >> 5) + 1)) * pbst, b1f);
        acc[0] = __builtin_amdgcn_wmma_f32_16x16x32_bf16(false, a0f, false, b0f[0], (short)0, acc[0], false, false);
        acc[1] = __builtin_amdgcn_wmma_f32_16x16x32_bf16(false, a0f, false, b0f[1], (short)0, acc[1], false, false);
        acc[2] = __builtin_amdgcn_wmma_f32_16x16x32_bf16(false, a0f, false, b0f[2], (short)0, acc[2], false, false);
        acc[3] = __builtin_amdgcn_wmma_f32_16x16x32_bf16(false, a0f, false, b0f[3], (short)0, acc[3], false, false);
        // stage 1: issue loads for k0+64, compute on set1 (k0+32)
        a0f = load_a_frag(arow + k0 + 64);
        load_b_frags(pb + ((size_t)((k0 >> 5) + 2)) * pbst, b0f);
        acc[0] = __builtin_amdgcn_wmma_f32_16x16x32_bf16(false, a1f, false, b1f[0], (short)0, acc[0], false, false);
        acc[1] = __builtin_amdgcn_wmma_f32_16x16x32_bf16(false, a1f, false, b1f[1], (short)0, acc[1], false, false);
        acc[2] = __builtin_amdgcn_wmma_f32_16x16x32_bf16(false, a1f, false, b1f[2], (short)0, acc[2], false, false);
        acc[3] = __builtin_amdgcn_wmma_f32_16x16x32_bf16(false, a1f, false, b1f[3], (short)0, acc[3], false, false);
    }
    // tail double-step: k0 == K-64
    a1f = load_a_frag(arow + k0 + 32);
    load_b_frags(pb + ((size_t)((k0 >> 5) + 1)) * pbst, b1f);
    acc[0] = __builtin_amdgcn_wmma_f32_16x16x32_bf16(false, a0f, false, b0f[0], (short)0, acc[0], false, false);
    acc[1] = __builtin_amdgcn_wmma_f32_16x16x32_bf16(false, a0f, false, b0f[1], (short)0, acc[1], false, false);
    acc[2] = __builtin_amdgcn_wmma_f32_16x16x32_bf16(false, a0f, false, b0f[2], (short)0, acc[2], false, false);
    acc[3] = __builtin_amdgcn_wmma_f32_16x16x32_bf16(false, a0f, false, b0f[3], (short)0, acc[3], false, false);
    acc[0] = __builtin_amdgcn_wmma_f32_16x16x32_bf16(false, a1f, false, b1f[0], (short)0, acc[0], false, false);
    acc[1] = __builtin_amdgcn_wmma_f32_16x16x32_bf16(false, a1f, false, b1f[1], (short)0, acc[1], false, false);
    acc[2] = __builtin_amdgcn_wmma_f32_16x16x32_bf16(false, a1f, false, b1f[2], (short)0, acc[2], false, false);
    acc[3] = __builtin_amdgcn_wmma_f32_16x16x32_bf16(false, a1f, false, b1f[3], (short)0, acc[3], false, false);

    // ---- epilogue: bias + relu + scatter to C ----
    const int mrow = tm * 16;
    #pragma unroll
    for (int nt = 0; nt < 4; ++nt) {
        const int n = tn0 + nt * 16 + mn;
        const float bv = bias ? bias[n] : 0.0f;
        #pragma unroll
        for (int r = 0; r < 8; ++r) {
            float v = acc[nt][r] + bv;
            if (relu) v = fmaxf(v, 0.0f);
            C[(size_t)(mrow + r + 8 * half) * N + n] = v;
        }
    }
}

// ---------------------------------------------------------------------------
// Scalar / atomic helper kernels (memory-bound sparse part)
// ---------------------------------------------------------------------------
__device__ __forceinline__ unsigned f2ord(float f) {
    unsigned u = __float_as_uint(f);
    return (u & 0x80000000u) ? ~u : (u | 0x80000000u);
}
__device__ __forceinline__ float ord2f(unsigned u) {
    return (u & 0x80000000u) ? __uint_as_float(u & 0x7FFFFFFFu)
                             : __uint_as_float(~u);
}

__global__ void fill_u32(unsigned* p, unsigned v, int n) {
    int i = blockIdx.x * blockDim.x + threadIdx.x;
    if (i < n) p[i] = v;
}

__global__ void deg_kernel(const int* __restrict__ dst, float* __restrict__ deg) {
    int e = blockIdx.x * blockDim.x + threadIdx.x;
    if (e >= ESL) return;
    int d = (e < NE) ? dst[e] : (e - NE);
    atomicAdd(&deg[d], 1.0f);
}

__global__ void dinv_kernel(const float* __restrict__ deg, float* __restrict__ dinv) {
    int i = blockIdx.x * blockDim.x + threadIdx.x;
    if (i >= NN) return;
    float d = deg[i];
    dinv[i] = (d > 0.0f) ? rsqrtf(d) : 0.0f;
}

// one wave per edge (incl self loops), lanes stride features
__global__ void gcn_agg(const float* __restrict__ h, const int* __restrict__ src,
                        const int* __restrict__ dst, const float* __restrict__ dinv,
                        float* __restrict__ out, int D)
{
    int gw   = (blockIdx.x * blockDim.x + threadIdx.x) >> 5;
    int lane = threadIdx.x & 31;
    if (gw >= ESL) return;
    int s, d;
    if (gw < NE) { s = src[gw]; d = dst[gw]; } else { s = d = gw - NE; }
    float norm = dinv[s] * dinv[d];
    const float* hs = h + (size_t)s * D;
    float* od = out + (size_t)d * D;
    for (int f = lane; f < D; f += 32)
        atomicAdd(&od[f], hs[f] * norm);
}

__global__ void bias_relu(float* __restrict__ h, const float* __restrict__ b, int D) {
    size_t i = (size_t)blockIdx.x * blockDim.x + threadIdx.x;
    if (i >= (size_t)NN * D) return;
    float v = h[i] + b[i % D];
    h[i] = fmaxf(v, 0.0f);
}

__global__ void gat_scores(const float* __restrict__ h, const float* __restrict__ att_s,
                           const float* __restrict__ att_d, float* __restrict__ a_src,
                           float* __restrict__ a_dst, int D)
{
    int n = blockIdx.x * blockDim.x + threadIdx.x;
    if (n >= NN) return;
    const float* hn = h + (size_t)n * D;
    float s = 0.0f, d = 0.0f;
    for (int k = 0; k < D; ++k) { float v = hn[k]; s += v * att_s[k]; d += v * att_d[k]; }
    a_src[n] = s; a_dst[n] = d;
}

__global__ void gat_edge_max(const float* __restrict__ a_src, const float* __restrict__ a_dst,
                             const int* __restrict__ src, const int* __restrict__ dst,
                             float* __restrict__ ebuf, unsigned* __restrict__ m_enc)
{
    int e = blockIdx.x * blockDim.x + threadIdx.x;
    if (e >= ESL) return;
    int s, d;
    if (e < NE) { s = src[e]; d = dst[e]; } else { s = d = e - NE; }
    float v = a_src[s] + a_dst[d];
    v = (v > 0.0f) ? v : 0.2f * v;            // leaky_relu(0.2)
    ebuf[e] = v;
    atomicMax(&m_enc[d], f2ord(v));
}

__global__ void gat_edge_exp(const int* __restrict__ dst, float* __restrict__ ebuf,
                             const unsigned* __restrict__ m_enc, float* __restrict__ ssum)
{
    int e = blockIdx.x * blockDim.x + threadIdx.x;
    if (e >= ESL) return;
    int d = (e < NE) ? dst[e] : (e - NE);
    float ex = __expf(ebuf[e] - ord2f(m_enc[d]));
    ebuf[e] = ex;
    atomicAdd(&ssum[d], ex);
}

__global__ void gat_edge_agg(const float* __restrict__ h, const int* __restrict__ src,
                             const int* __restrict__ dst, const float* __restrict__ ebuf,
                             const float* __restrict__ ssum, float* __restrict__ out, int D)
{
    int gw   = (blockIdx.x * blockDim.x + threadIdx.x) >> 5;
    int lane = threadIdx.x & 31;
    if (gw >= ESL) return;
    int s, d;
    if (gw < NE) { s = src[gw]; d = dst[gw]; } else { s = d = gw - NE; }
    float alpha = ebuf[gw] / ssum[d];
    const float* hs = h + (size_t)s * D;
    float* od = out + (size_t)d * D;
    for (int f = lane; f < D; f += 32)
        atomicAdd(&od[f], hs[f] * alpha);
}

// GIN sum aggregation over real edges only
__global__ void gin_agg(const float* __restrict__ h, const int* __restrict__ src,
                        const int* __restrict__ dst, float* __restrict__ out, int D)
{
    int gw   = (blockIdx.x * blockDim.x + threadIdx.x) >> 5;
    int lane = threadIdx.x & 31;
    if (gw >= NE) return;
    int s = src[gw], d = dst[gw];
    const float* hs = h + (size_t)s * D;
    float* od = out + (size_t)d * D;
    for (int f = lane; f < D; f += 32)
        atomicAdd(&od[f], hs[f]);
}

__global__ void add_inplace(float* __restrict__ a, const float* __restrict__ b, size_t n) {
    size_t i = (size_t)blockIdx.x * blockDim.x + threadIdx.x;
    if (i < n) a[i] += b[i];
}

__global__ void pool_sum(const float* __restrict__ h, const int* __restrict__ batch,
                         float* __restrict__ psum, int D)
{
    int gw   = (blockIdx.x * blockDim.x + threadIdx.x) >> 5;
    int lane = threadIdx.x & 31;
    if (gw >= NN) return;
    int g = batch[gw];
    const float* hn = h + (size_t)gw * D;
    float* pg = psum + (size_t)g * D;
    for (int f = lane; f < D; f += 32)
        atomicAdd(&pg[f], hn[f]);
}

__global__ void pool_cnt(const int* __restrict__ batch, float* __restrict__ cnt) {
    int n = blockIdx.x * blockDim.x + threadIdx.x;
    if (n >= NN) return;
    atomicAdd(&cnt[batch[n]], 1.0f);
}

__global__ void final_linear(const float* __restrict__ psum, const float* __restrict__ cnt,
                             const float* __restrict__ W, const float* __restrict__ b,
                             float* __restrict__ out)
{
    int i = blockIdx.x * blockDim.x + threadIdx.x;
    if (i >= NG * NY) return;
    int g = i / NY, y = i % NY;
    const float* pg = psum + (size_t)g * DMAX;
    float s = 0.0f;
    for (int d = 0; d < DMAX; ++d) s += pg[d] * W[d * NY + y];
    float c = cnt[g]; c = (c > 1.0f) ? c : 1.0f;
    out[i] = s / c + b[y];   // (sums @ W)/cnt == (sums/cnt) @ W
}

// ---------------------------------------------------------------------------
// Launcher
// ---------------------------------------------------------------------------
static inline int cdiv(long long a, long long b) { return (int)((a + b - 1) / b); }

extern "C" void kernel_launch(void* const* d_in, const int* in_sizes, int n_in,
                              void* d_out, int out_size, void* d_ws, size_t ws_size,
                              hipStream_t stream)
{
    (void)in_sizes; (void)n_in; (void)out_size; (void)ws_size;

    const float* x       = (const float*)d_in[0];
    const int*   eidx    = (const int*)d_in[1];
    const int*   batch   = (const int*)d_in[2];
    const float* gcn1_w  = (const float*)d_in[3];
    const float* gcn1_b  = (const float*)d_in[4];
    const float* gcn2_w  = (const float*)d_in[5];
    const float* gcn2_b  = (const float*)d_in[6];
    const float* gat1_w  = (const float*)d_in[7];
    const float* gat1_as = (const float*)d_in[8];
    const float* gat1_ad = (const float*)d_in[9];
    const float* gat1_b  = (const float*)d_in[10];
    const float* gat2_w  = (const float*)d_in[11];
    const float* gat2_as = (const float*)d_in[12];
    const float* gat2_ad = (const float*)d_in[13];
    const float* gat2_b  = (const float*)d_in[14];
    const float* gin_w1  = (const float*)d_in[15];
    const float* gin_b1  = (const float*)d_in[16];
    const float* gin_w2  = (const float*)d_in[17];
    const float* gin_b2  = (const float*)d_in[18];
    const float* out_w   = (const float*)d_in[19];
    const float* out_b   = (const float*)d_in[20];
    float* out = (float*)d_out;

    const int* src = eidx;          // edge_index[0]
    const int* dst = eidx + NE;     // edge_index[1]

    // ---- workspace carve-up ----
    float* ws = (float*)d_ws;
    float*    deg   = ws;                 ws += NN;
    float*    dinv  = ws;                 ws += NN;
    float*    a_src = ws;                 ws += NN;
    float*    a_dst = ws;                 ws += NN;
    unsigned* m_enc = (unsigned*)ws;      ws += NN;
    float*    ssum  = ws;                 ws += NN;
    float*    cnt   = ws;                 ws += NG;
    float*    psum  = ws;                 ws += (size_t)NG * DMAX;
    float*    ebuf  = ws;                 ws += ESL;

    // packed bf16 weights (1 MB total), 256B aligned
    uintptr_t p = (uintptr_t)ws; p = (p + 255) & ~(uintptr_t)255;
    __bf16* pw_gcn1 = (__bf16*)p;
    __bf16* pw_gcn2 = pw_gcn1 + 128 * 128;
    __bf16* pw_gat1 = pw_gcn2 + 128 * 128;
    __bf16* pw_gat2 = pw_gat1 + 128 * 256;
    __bf16* pw_gin1 = pw_gat2 + 256 * 256;
    __bf16* pw_gin2 = pw_gin1 + 256 * 512;
    __bf16* pw_end  = pw_gin2 + 512 * 512;

    // big ping-pong buffers, 256B aligned
    p = (uintptr_t)pw_end; p = (p + 255) & ~(uintptr_t)255;
    float* bufA = (float*)p;
    float* bufB = bufA + (size_t)NN * DMAX;

    const int T = 256;
    const unsigned NEG_INF_ORD = 0x007FFFFFu;   // f2ord(-inf)
    #define EDGE_GRID   cdiv(ESL, T)
    #define EWAVE_GRID  cdiv((long long)ESL * 32, T)
    #define NODE_GRID   cdiv(NN, T)
    // GEMM grid: ((M/16)*(N/64))/4 blocks of 128 threads
    #define GEMM_GRID(Ncols) ((NN / 16) * ((Ncols) / 64) / 4)

    // ---- pre-pack all weight matrices to bf16 fragment order ----
    pack_w_bf16<<<cdiv((128/32)*(128/16)*32, T), T, 0, stream>>>(gcn1_w, pw_gcn1, 128, 128);
    pack_w_bf16<<<cdiv((128/32)*(128/16)*32, T), T, 0, stream>>>(gcn2_w, pw_gcn2, 128, 128);
    pack_w_bf16<<<cdiv((128/32)*(256/16)*32, T), T, 0, stream>>>(gat1_w, pw_gat1, 128, 256);
    pack_w_bf16<<<cdiv((256/32)*(256/16)*32, T), T, 0, stream>>>(gat2_w, pw_gat2, 256, 256);
    pack_w_bf16<<<cdiv((256/32)*(512/16)*32, T), T, 0, stream>>>(gin_w1, pw_gin1, 256, 512);
    pack_w_bf16<<<cdiv((512/32)*(512/16)*32, T), T, 0, stream>>>(gin_w2, pw_gin2, 512, 512);

    // ---- degrees (with self loops) + rsqrt ----
    hipMemsetAsync(deg, 0, NN * sizeof(float), stream);
    deg_kernel<<<EDGE_GRID, T, 0, stream>>>(dst, deg);
    dinv_kernel<<<NODE_GRID, T, 0, stream>>>(deg, dinv);

    // ---- GCN1: lin -> agg -> bias+relu  (D=128) ----
    wmma_gemm_bf16<<<GEMM_GRID(128), 128, 0, stream>>>(x, pw_gcn1, nullptr, bufA, NN, 128, 128, 0);
    hipMemsetAsync(bufB, 0, (size_t)NN * 128 * sizeof(float), stream);
    gcn_agg<<<EWAVE_GRID, T, 0, stream>>>(bufA, src, dst, dinv, bufB, 128);
    bias_relu<<<cdiv((long long)NN * 128, T), T, 0, stream>>>(bufB, gcn1_b, 128);

    // ---- GCN2 (D=128) ----
    wmma_gemm_bf16<<<GEMM_GRID(128), 128, 0, stream>>>(bufB, pw_gcn2, nullptr, bufA, NN, 128, 128, 0);
    hipMemsetAsync(bufB, 0, (size_t)NN * 128 * sizeof(float), stream);
    gcn_agg<<<EWAVE_GRID, T, 0, stream>>>(bufA, src, dst, dinv, bufB, 128);
    bias_relu<<<cdiv((long long)NN * 128, T), T, 0, stream>>>(bufB, gcn2_b, 128);

    // ---- GAT1: lin (128->256), edge softmax, agg, bias+relu ----
    wmma_gemm_bf16<<<GEMM_GRID(256), 128, 0, stream>>>(bufB, pw_gat1, nullptr, bufA, NN, 128, 256, 0);
    gat_scores<<<NODE_GRID, T, 0, stream>>>(bufA, gat1_as, gat1_ad, a_src, a_dst, 256);
    fill_u32<<<NODE_GRID, T, 0, stream>>>(m_enc, NEG_INF_ORD, NN);
    hipMemsetAsync(ssum, 0, NN * sizeof(float), stream);
    gat_edge_max<<<EDGE_GRID, T, 0, stream>>>(a_src, a_dst, src, dst, ebuf, m_enc);
    gat_edge_exp<<<EDGE_GRID, T, 0, stream>>>(dst, ebuf, m_enc, ssum);
    hipMemsetAsync(bufB, 0, (size_t)NN * 256 * sizeof(float), stream);
    gat_edge_agg<<<EWAVE_GRID, T, 0, stream>>>(bufA, src, dst, ebuf, ssum, bufB, 256);
    bias_relu<<<cdiv((long long)NN * 256, T), T, 0, stream>>>(bufB, gat1_b, 256);

    // ---- GAT2 (256->256) ----
    wmma_gemm_bf16<<<GEMM_GRID(256), 128, 0, stream>>>(bufB, pw_gat2, nullptr, bufA, NN, 256, 256, 0);
    gat_scores<<<NODE_GRID, T, 0, stream>>>(bufA, gat2_as, gat2_ad, a_src, a_dst, 256);
    fill_u32<<<NODE_GRID, T, 0, stream>>>(m_enc, NEG_INF_ORD, NN);
    hipMemsetAsync(ssum, 0, NN * sizeof(float), stream);
    gat_edge_max<<<EDGE_GRID, T, 0, stream>>>(a_src, a_dst, src, dst, ebuf, m_enc);
    gat_edge_exp<<<EDGE_GRID, T, 0, stream>>>(dst, ebuf, m_enc, ssum);
    hipMemsetAsync(bufB, 0, (size_t)NN * 256 * sizeof(float), stream);
    gat_edge_agg<<<EWAVE_GRID, T, 0, stream>>>(bufA, src, dst, ebuf, ssum, bufB, 256);
    bias_relu<<<cdiv((long long)NN * 256, T), T, 0, stream>>>(bufB, gat2_b, 256);

    // ---- GIN: agg(real edges) + x, MLP(256->512->512) with relu ----
    hipMemsetAsync(bufA, 0, (size_t)NN * 256 * sizeof(float), stream);
    gin_agg<<<cdiv((long long)NE * 32, T), T, 0, stream>>>(bufB, src, dst, bufA, 256);
    add_inplace<<<cdiv((long long)NN * 256, T), T, 0, stream>>>(bufA, bufB, (size_t)NN * 256);
    wmma_gemm_bf16<<<GEMM_GRID(512), 128, 0, stream>>>(bufA, pw_gin1, gin_b1, bufB, NN, 256, 512, 1);
    wmma_gemm_bf16<<<GEMM_GRID(512), 128, 0, stream>>>(bufB, pw_gin2, gin_b2, bufA, NN, 512, 512, 1);

    // ---- mean pool + output linear ----
    hipMemsetAsync(psum, 0, (size_t)NG * DMAX * sizeof(float), stream);
    hipMemsetAsync(cnt, 0, NG * sizeof(float), stream);
    pool_sum<<<cdiv((long long)NN * 32, T), T, 0, stream>>>(bufA, batch, psum, 512);
    pool_cnt<<<NODE_GRID, T, 0, stream>>>(batch, cnt);
    final_linear<<<cdiv(NG * NY, 64), 64, 0, stream>>>(psum, cnt, out_w, out_b, out);
}
